// TransAttention_27728308863564
// MI455X (gfx1250) — matrix operand order repaired and checked by reading, not compiled, not run on hardware
//
#include <hip/hip_runtime.h>
#include <hip/hip_bf16.h>
#include <math.h>

typedef __attribute__((ext_vector_type(16))) _Float16 v16h;
typedef __attribute__((ext_vector_type(8)))  float    v8f;
typedef __attribute__((ext_vector_type(4)))  int      i32x4;
typedef unsigned long long u64;

// pointer-to-vector in explicit address spaces for the async-DMA builtin
typedef __attribute__((address_space(1))) i32x4* gptr128;   // global
typedef __attribute__((address_space(3))) i32x4* lptr128;   // LDS

#define C_DIM   192
#define C3_DIM  576
#define HW_DIM  65536
#define NSEG    8192     // spatial per head-segment
#define HEADS   8
#define BATCH   2

// ---------------------------------------------------------------------------
// qkv pointwise GEMM:  Y[b, m, p] = sum_k W[m,k] * X[b, k, p]   (K = 192, f32 in)
// Block = 128 threads (4 waves) -> 16(M) x 64(N) tile, K stepped by 32.
// Staging is two-phase: 4 coalesced float4 loads per thread into registers,
// then cvt+scatter into the transposed f16 LDS tile (so B-fragments are
// contiguous 16-half runs in this code's k order: lane = column, k = 16*half + i).
// ---------------------------------------------------------------------------
__global__ void gemm_qkv(const float* __restrict__ W, const float* __restrict__ X,
                         float* __restrict__ Y) {
  __shared__ _Float16 xs[64][40];   // [n][k], padded
  const int n0 = blockIdx.x * 64;
  const int m0 = blockIdx.y * 16;
  const float* Xb = X + (size_t)blockIdx.z * (C_DIM * HW_DIM);
  float*       Yb = Y + (size_t)blockIdx.z * ((size_t)C3_DIM * HW_DIM);

  const int t    = threadIdx.x;
  const int wave = t >> 5;
  const int lane = t & 31;
  const int half = lane >> 4;
  const int colw = n0 + wave * 16 + (lane & 15);

  v8f acc = {};
#pragma unroll
  for (int kk = 0; kk < C_DIM; kk += 32) {
    __syncthreads();
    float4 r[4];
#pragma unroll
    for (int i = 0; i < 4; ++i) {
      int idx = i * 128 + t;              // 0..511 quads
      int k   = idx >> 4;                 // 0..31
      int n4  = (idx & 15) << 2;          // 0..60
      r[i] = *(const float4*)(Xb + ((kk + k) << 16) + n0 + n4);
      if (kk + 32 < C_DIM)
        __builtin_prefetch(Xb + ((kk + 32 + k) << 16) + n0 + n4, 0, 0);
    }
#pragma unroll
    for (int i = 0; i < 4; ++i) {
      int idx = i * 128 + t;
      int k = idx >> 4, n4 = (idx & 15) << 2;
      xs[n4 + 0][k] = (_Float16)r[i].x;
      xs[n4 + 1][k] = (_Float16)r[i].y;
      xs[n4 + 2][k] = (_Float16)r[i].z;
      xs[n4 + 3][k] = (_Float16)r[i].w;
    }
    __syncthreads();

    v16h a, b;
    const float* wr = W + (m0 + (lane & 15)) * C_DIM + kk;
#pragma unroll
    for (int i = 0; i < 8; ++i) {
      a[i]     = (_Float16)wr[16 * half + i];
      a[8 + i] = (_Float16)wr[16 * half + 8 + i];
    }
    const _Float16* bp = &xs[wave * 16 + (lane & 15)][16 * half];
#pragma unroll
    for (int i = 0; i < 16; ++i) b[i] = bp[i];

    acc = __builtin_amdgcn_wmma_f32_16x16x32_f16(false, a, false, b,
                                                 (short)0, acc, false, false);
  }
#pragma unroll
  for (int r = 0; r < 8; ++r)
    Yb[(size_t)((m0 + r + 8 * half) << 16) + colw] = acc[r];
}

// ---------------------------------------------------------------------------
// Depthwise 3x3, SAME zero padding. Pure bandwidth.
// ---------------------------------------------------------------------------
__global__ void dwconv3x3(const float* __restrict__ in, const float* __restrict__ w,
                          float* __restrict__ out) {
  size_t idx = (size_t)blockIdx.x * 256 + threadIdx.x;   // b*576*65536 total
  int x = (int)(idx & 255);
  int y = (int)((idx >> 8) & 255);
  size_t plane = idx >> 16;                  // b*576 + ch
  int ch = (int)(plane % C3_DIM);
  const float* wp = w + ch * 9;
  const float* ip = in + (plane << 16);
  float s = 0.f;
#pragma unroll
  for (int dy = -1; dy <= 1; ++dy) {
#pragma unroll
    for (int dx = -1; dx <= 1; ++dx) {
      int yy = y + dy, xx = x + dx;
      if (yy >= 0 && yy < 256 && xx >= 0 && xx < 256)
        s = fmaf(wp[(dy + 1) * 3 + (dx + 1)], ip[yy * 256 + xx], s);
    }
  }
  out[idx] = s;
}

// ---------------------------------------------------------------------------
// L2-normalize q,k rows (each row = contiguous 8192-float segment), emit f16.
// ---------------------------------------------------------------------------
__global__ void l2norm_f16(const float* __restrict__ qkv_dw,
                           _Float16* __restrict__ qn, _Float16* __restrict__ kn) {
  const int r    = blockIdx.x;
  const int part = r / (BATCH * C_DIM * 8);        // 0 = q, 1 = k
  int rem  = r % (BATCH * C_DIM * 8);
  const int b   = rem / (C_DIM * 8);
  rem %= (C_DIM * 8);
  const int ch  = rem >> 3;
  const int seg = rem & 7;

  const float* src = qkv_dw + (((size_t)(b * C3_DIM + part * C_DIM + ch)) << 16) + (seg << 13);
  _Float16*    dst = (part ? kn : qn) + (((size_t)(b * C_DIM + ch)) << 16) + (seg << 13);

  __shared__ float red[8];
  __shared__ float s_scale;
  const int t = threadIdx.x;
  float ss = 0.f;
  for (int i = t; i < NSEG; i += 256) { float v = src[i]; ss = fmaf(v, v, ss); }
#pragma unroll
  for (int off = 16; off; off >>= 1) ss += __shfl_xor(ss, off);
  if ((t & 31) == 0) red[t >> 5] = ss;
  __syncthreads();
  if (t == 0) {
    float tot = 0.f;
#pragma unroll
    for (int i = 0; i < 8; ++i) tot += red[i];
    s_scale = 1.0f / fmaxf(sqrtf(tot), 1e-12f);
  }
  __syncthreads();
  const float sc = s_scale;
  for (int i = t; i < NSEG; i += 256) dst[i] = (_Float16)(src[i] * sc);
}

// ---------------------------------------------------------------------------
// attn[b,h,i,j] = temperature[h] * sum_n qn_row_i[n] * kn_row_j[n]
// M=N=192, K=8192. A and B fragments are contiguous f16 row reads (2x b128).
// ---------------------------------------------------------------------------
__global__ void qk_attn(const _Float16* __restrict__ qn, const _Float16* __restrict__ kn,
                        const float* __restrict__ temp, float* __restrict__ attn) {
  const int bh = blockIdx.y, b = bh >> 3, head = bh & 7;
  const int wave = threadIdx.x >> 5, lane = threadIdx.x & 31, half = lane >> 4;
  const int tile = blockIdx.x * 8 + wave;      // 0..143
  const int m0 = (tile / 12) * 16, j0 = (tile % 12) * 16;

  const int m = m0 + (lane & 15);
  const int j = j0 + (lane & 15);
  const _Float16* ar = qn + (((size_t)(b * C_DIM + head * 24 + (m >> 3))) << 16) + ((m & 7) << 13);
  const _Float16* br = kn + (((size_t)(b * C_DIM + head * 24 + (j >> 3))) << 16) + ((j & 7) << 13);

  v8f acc = {};
  for (int kk = 0; kk < NSEG; kk += 32) {
    v16h a, bb;
#pragma unroll
    for (int i = 0; i < 8; ++i) {
      a[i]     = ar[kk + 16 * half + i];
      a[8 + i] = ar[kk + 16 * half + 8 + i];
    }
#pragma unroll
    for (int i = 0; i < 16; ++i) bb[i] = br[kk + 16 * half + i];
    acc = __builtin_amdgcn_wmma_f32_16x16x32_f16(false, a, false, bb,
                                                 (short)0, acc, false, false);
  }
  const float T = temp[head];
  float* out = attn + (size_t)bh * C_DIM * C_DIM;
#pragma unroll
  for (int r = 0; r < 8; ++r)
    out[(m0 + r + 8 * half) * C_DIM + j0 + (lane & 15)] = acc[r] * T;
}

// ---------------------------------------------------------------------------
// Row softmax over 192, emit f16. One block (192 threads = 6 waves) per row.
// ---------------------------------------------------------------------------
__global__ void softmax192(const float* __restrict__ attn, _Float16* __restrict__ attn_h) {
  const int row = blockIdx.x, t = threadIdx.x;
  __shared__ float sred[6];
  const float v = attn[(size_t)row * C_DIM + t];

  float m = v;
#pragma unroll
  for (int off = 16; off; off >>= 1) m = fmaxf(m, __shfl_xor(m, off));
  if ((t & 31) == 0) sred[t >> 5] = m;
  __syncthreads();
  float mm = sred[0];
#pragma unroll
  for (int i = 1; i < 6; ++i) mm = fmaxf(mm, sred[i]);
  __syncthreads();

  const float e = __expf(v - mm);
  float s = e;
#pragma unroll
  for (int off = 16; off; off >>= 1) s += __shfl_xor(s, off);
  if ((t & 31) == 0) sred[t >> 5] = s;
  __syncthreads();
  float tot = sred[0];
#pragma unroll
  for (int i = 1; i < 6; ++i) tot += sred[i];

  attn_h[(size_t)row * C_DIM + t] = (_Float16)(e / tot);
}

// ---------------------------------------------------------------------------
// out[b,h,m,n] = sum_d attn[b,h,m,d] * v_row(b,h,d)[n]   (K = 192), f16 output.
// v staged two-phase (coalesced float4 -> registers -> cvt -> LDS transpose).
// Output stored flat at (head*192+m)*8192+n == the (c', spatial') layout.
// ---------------------------------------------------------------------------
__global__ void av_gemm(const _Float16* __restrict__ attn_h,
                        const float* __restrict__ qkv_dw, _Float16* __restrict__ oa) {
  __shared__ _Float16 vs[64][40];
  const int bh = blockIdx.z, b = bh >> 3, head = bh & 7;
  const int n0 = blockIdx.x * 64;
  const int m0 = blockIdx.y * 16;
  const int t = threadIdx.x, wave = t >> 5, lane = t & 31, half = lane >> 4;
  const _Float16* A = attn_h + (size_t)bh * C_DIM * C_DIM;

  v8f acc = {};
#pragma unroll
  for (int kk = 0; kk < C_DIM; kk += 32) {
    __syncthreads();
    float4 r[4];
#pragma unroll
    for (int i = 0; i < 4; ++i) {
      int idx = i * 128 + t;
      int k = idx >> 4, n4 = (idx & 15) << 2;
      int d = kk + k;
      int off = ((b * C3_DIM + 384 + head * 24 + (d >> 3)) << 16) + ((d & 7) << 13) + n0 + n4;
      r[i] = *(const float4*)(qkv_dw + off);
    }
#pragma unroll
    for (int i = 0; i < 4; ++i) {
      int idx = i * 128 + t;
      int k = idx >> 4, n4 = (idx & 15) << 2;
      vs[n4 + 0][k] = (_Float16)r[i].x;
      vs[n4 + 1][k] = (_Float16)r[i].y;
      vs[n4 + 2][k] = (_Float16)r[i].z;
      vs[n4 + 3][k] = (_Float16)r[i].w;
    }
    __syncthreads();

    v16h a, bb;
    const _Float16* ar = A + (m0 + (lane & 15)) * C_DIM + kk;
#pragma unroll
    for (int i = 0; i < 8; ++i) {
      a[i]     = ar[16 * half + i];
      a[8 + i] = ar[16 * half + 8 + i];
    }
    const _Float16* bp = &vs[wave * 16 + (lane & 15)][16 * half];
#pragma unroll
    for (int i = 0; i < 16; ++i) bb[i] = bp[i];

    acc = __builtin_amdgcn_wmma_f32_16x16x32_f16(false, a, false, bb,
                                                 (short)0, acc, false, false);
  }
  _Float16* ob = oa + (size_t)b * (C_DIM * HW_DIM) + head * (C_DIM * NSEG);
  const int col = n0 + wave * 16 + (lane & 15);
#pragma unroll
  for (int r = 0; r < 8; ++r)
    ob[(m0 + r + 8 * half) * NSEG + col] = (_Float16)acc[r];
}

// ---------------------------------------------------------------------------
// Final projection: Y[b,m,p] = sum_k proj_w[m,k] * OA[b,k,p]  (f16 input).
// Input staging is a pure byte copy -> GLOBAL_LOAD_ASYNC_TO_LDS_B128
// (ASYNCcnt-tracked) when the builtin exists; fallback = vector register copy.
// LDS tile is [k][n] row-major (async DMA can't transpose), so B-fragments
// are per-lane strided u16 reads; A-fragments come from the f32 weights.
// ---------------------------------------------------------------------------
__global__ void gemm_proj(const float* __restrict__ W, const _Float16* __restrict__ X,
                          float* __restrict__ Y) {
  __shared__ _Float16 xs[32][72];   // [k][n], 144B row stride (16B aligned)
  const int n0 = blockIdx.x * 64;
  const int m0 = blockIdx.y * 16;
  const _Float16* Xb = X + (size_t)blockIdx.z * (C_DIM * HW_DIM);
  float*          Yb = Y + (size_t)blockIdx.z * (C_DIM * HW_DIM);

  const int t = threadIdx.x, wave = t >> 5, lane = t & 31, half = lane >> 4;
  const int colw = n0 + wave * 16 + (lane & 15);

  v8f acc = {};
#pragma unroll
  for (int kk = 0; kk < C_DIM; kk += 32) {
    __syncthreads();
#pragma unroll
    for (int i = 0; i < 2; ++i) {
      int idx = i * 128 + t;              // 0..255
      int k   = idx >> 3;                 // 0..31
      int n8  = (idx & 7) << 3;           // 0..56 (16B granules)
      const _Float16* src = Xb + ((kk + k) << 16) + n0 + n8;
      _Float16*       dst = &xs[k][n8];
#if __has_builtin(__builtin_amdgcn_global_load_async_to_lds_b128)
      __builtin_amdgcn_global_load_async_to_lds_b128(
          (gptr128)src, (lptr128)dst, 0, 0);
#else
      ((u64*)dst)[0] = ((const u64*)src)[0];
      ((u64*)dst)[1] = ((const u64*)src)[1];
#endif
    }
#if __has_builtin(__builtin_amdgcn_s_wait_asynccnt)
    __builtin_amdgcn_s_wait_asynccnt(0);
#else
    asm volatile("s_wait_asynccnt 0x0" ::: "memory");
#endif
    __syncthreads();

    v16h a, bb;
    const float* wr = W + (m0 + (lane & 15)) * C_DIM + kk;
#pragma unroll
    for (int i = 0; i < 8; ++i) {
      a[i]     = (_Float16)wr[16 * half + i];
      a[8 + i] = (_Float16)wr[16 * half + 8 + i];
    }
    const int col = wave * 16 + (lane & 15);
#pragma unroll
    for (int i = 0; i < 16; ++i) bb[i] = xs[16 * half + i][col];

    acc = __builtin_amdgcn_wmma_f32_16x16x32_f16(false, a, false, bb,
                                                 (short)0, acc, false, false);
  }
#pragma unroll
  for (int r = 0; r < 8; ++r)
    Yb[(size_t)((m0 + r + 8 * half) << 16) + colw] = acc[r];
}

// ---------------------------------------------------------------------------
extern "C" void kernel_launch(void* const* d_in, const int* in_sizes, int n_in,
                              void* d_out, int out_size, void* d_ws, size_t ws_size,
                              hipStream_t stream) {
  const float* x      = (const float*)d_in[0];  // (2,192,256,256)
  const float* qkv_w  = (const float*)d_in[1];  // (576,192)
  const float* dw_w   = (const float*)d_in[2];  // (576,1,3,3)
  const float* proj_w = (const float*)d_in[3];  // (192,192)
  const float* temp   = (const float*)d_in[4];  // (8,1,1)

  char* ws = (char*)d_ws;
  const size_t QKV_F = (size_t)BATCH * C3_DIM * HW_DIM;          // 75,497,472
  const size_t QK_H  = (size_t)BATCH * C_DIM * HW_DIM;           // 25,165,824
  const size_t ATT_F = (size_t)BATCH * HEADS * C_DIM * C_DIM;    // 589,824

  float*    qkv    = (float*)ws;       ws += QKV_F * 4;
  float*    qkv_dw = (float*)ws;       ws += QKV_F * 4;
  _Float16* qn     = (_Float16*)ws;    ws += QK_H * 2;
  _Float16* kn     = (_Float16*)ws;    ws += QK_H * 2;
  float*    attn   = (float*)ws;       ws += ATT_F * 4;
  _Float16* attn_h = (_Float16*)ws;    ws += ATT_F * 2;
  _Float16* oa     = (_Float16*)ws;    // 50 MB, f16

  // 1) qkv = W_qkv x X   (M=576, K=192)
  gemm_qkv<<<dim3(HW_DIM / 64, C3_DIM / 16, BATCH), 128, 0, stream>>>(qkv_w, x, qkv);

  // 2) depthwise 3x3
  dwconv3x3<<<(unsigned)(QKV_F / 256), 256, 0, stream>>>(qkv, dw_w, qkv_dw);

  // 3) L2-normalize q,k rows -> f16
  l2norm_f16<<<2 * BATCH * C_DIM * 8, 256, 0, stream>>>(qkv_dw, qn, kn);

  // 4) attn = (q . k^T) * temperature   (per b,head: 192x192, K=8192)
  qk_attn<<<dim3(18, BATCH * HEADS), 256, 0, stream>>>(qn, kn, temp, attn);

  // 5) softmax over last dim (192), emit f16
  softmax192<<<BATCH * HEADS * C_DIM, C_DIM, 0, stream>>>(attn, attn_h);

  // 6) out = attn x v   (per b,head: 192x8192, K=192) -> f16
  av_gemm<<<dim3(NSEG / 64, C_DIM / 16, BATCH * HEADS), 128, 0, stream>>>(
      attn_h, qkv_dw, oa);

  // 7) final projection (M=192, K=192) -> d_out (f32)
  gemm_proj<<<dim3(HW_DIM / 64, C_DIM / 16, BATCH), 128, 0, stream>>>(
      proj_w, oa, (float*)d_out);
}
